// DrugRepurposingGNN_44324062495025
// MI455X (gfx1250) — matrix operand order repaired and checked by reading, not compiled, not run on hardware
//
#include <hip/hip_runtime.h>
#include <hip/hip_bf16.h>
#include <math.h>

// ---------------------------------------------------------------------------
// MI455X (gfx1250) heterogeneous GraphSAGE forward.
// Bottleneck is the edge gather/scatter (~4.6 GB/layer @ 23.3 TB/s HBM); all
// dense ops use v_wmma_f32_16x16x32_bf16 (bf16 in, fp32 accum). LDS tiles are
// stored fragment-contiguous so A and B WMMA operands load as ds_load_b128.
// ---------------------------------------------------------------------------

typedef __bf16 bf16;
typedef __attribute__((ext_vector_type(16))) __bf16 v16bf;
typedef __attribute__((ext_vector_type(8)))  float  v8f;

__device__ __forceinline__ unsigned short f2bfu(float f) {
  unsigned u = __builtin_bit_cast(unsigned, f);
  u += 0x7FFFu + ((u >> 16) & 1u);            // round-to-nearest-even
  return (unsigned short)(u >> 16);
}

// ---------------------------------------------------------------------------
__global__ void zero_f32(float* __restrict__ p, long long n) {
  long long i = (long long)blockIdx.x * blockDim.x + threadIdx.x;
  long long stride = (long long)gridDim.x * blockDim.x;
  for (; i < n; i += stride) p[i] = 0.0f;
}

// Per-edge-type in-degree (layer-invariant; computed once, reused 3x).
__global__ __launch_bounds__(256) void edge_count(
    const int* __restrict__ dst, float* __restrict__ cnt, int E) {
  int e = blockIdx.x * blockDim.x + threadIdx.x;
  if (e < E) atomicAdd(cnt + dst[e], 1.0f);
}

// One 32-lane group per edge: gather h_src row (float4/lane), scatter-add
// into agg[dst].
__global__ __launch_bounds__(256) void sage_scatter(
    const float* __restrict__ hsrc, const int* __restrict__ src,
    const int* __restrict__ dst, float* __restrict__ agg, int E) {
  int t = blockIdx.x * blockDim.x + threadIdx.x;
  int e = t >> 5;
  if (e >= E) return;
  int lane = t & 31;
  int s = src[e], d = dst[e];
  const float4 v = ((const float4*)(hsrc + (size_t)s * 128))[lane];
  float* a = agg + (size_t)d * 128 + lane * 4;
  atomicAdd(a + 0, v.x);
  atomicAdd(a + 1, v.y);
  atomicAdd(a + 2, v.z);
  atomicAdd(a + 3, v.w);
}

// ---------------------------------------------------------------------------
// LDS tile layouts (bf16 pairs packed in b32):
//   As32[r*16 + kp]  : A tile, row-major 16x32       -> a-frag = 2x ds_load_b128
//   Bs32[c*16 + kp]  : B tile, column-major 128x32   -> b-frag = 2x ds_load_b128
// Lane mapping used here (16-bit operands, wave32; contiguous k, not AMD's table order):
//   A: lane l holds row (l&15); K = kb + e,              kb = (l<16)?0:16
//   B: lane l holds col (l&15); K = kb + e,              kb = (l<16)?0:16
//   C/D: elem r -> row r + ((l<16)?0:8), col (l&15) per wave tile.
// ---------------------------------------------------------------------------
__global__ __launch_bounds__(256) void gemm_bf16_128(
    const float* __restrict__ A, const float* __restrict__ B,
    const float* __restrict__ bias, float* __restrict__ C,
    int K, int do_relu) {
  __shared__ unsigned As32[16 * 16];
  __shared__ unsigned Bs32[128 * 16];
  const bf16* Asb = (const bf16*)As32;
  const bf16* Bsb = (const bf16*)Bs32;
  const int tid  = threadIdx.x;
  const int lane = tid & 31;
  const int wave = tid >> 5;
  const long long row0 = (long long)blockIdx.x * 16;
  const int am  = lane & 15;
  const int akb = (lane < 16) ? 0 : 16;
  const int bkb = (lane < 16) ? 0 : 16;
  const int col = (wave << 4) + (lane & 15);
  v8f acc = {};
  for (int k0 = 0; k0 < K; k0 += 32) {
    if (k0 + 32 < K)
      __builtin_prefetch(&A[(row0 + (tid >> 4)) * K + k0 + 32], 0, 1);
    {
      int r = tid >> 4, kp = tid & 15;               // A: one bf16-pair/thread
      float2 f = *(const float2*)&A[(row0 + r) * K + k0 + 2 * kp];
      As32[r * 16 + kp] = (unsigned)f2bfu(f.x) | ((unsigned)f2bfu(f.y) << 16);
    }
#pragma unroll
    for (int i = 0; i < 8; ++i) {                    // B: 2048 bf16-pairs
      int idx = tid + i * 256;
      int kh = idx >> 7, c = idx & 127;
      float f0 = B[(long long)(k0 + 2 * kh) * 128 + c];
      float f1 = B[(long long)(k0 + 2 * kh + 1) * 128 + c];
      Bs32[c * 16 + kh] = (unsigned)f2bfu(f0) | ((unsigned)f2bfu(f1) << 16);
    }
    __syncthreads();
    v16bf a, b;
#pragma unroll
    for (int e = 0; e < 16; ++e) {
      a[e] = Asb[am * 32 + akb + e];
      b[e] = Bsb[col * 32 + bkb + e];
    }
    acc = __builtin_amdgcn_wmma_f32_16x16x32_bf16(false, a, false, b,
                                                  (short)0, acc, false, false);
    __syncthreads();
  }
  const int mofs = (lane < 16) ? 0 : 8;
  const float bv = bias ? bias[col] : 0.0f;
#pragma unroll
  for (int r = 0; r < 8; ++r) {
    float v = acc[r] + bv;
    if (do_relu) v = fmaxf(v, 0.0f);
    C[(row0 + r + mofs) * 128 + col] = v;
  }
}

// ---------------------------------------------------------------------------
// Fused SAGE conv: out = L2norm((agg/cnt)@Wl + bl + h_dst@Wr);
// hsum = out (accum=0) or hsum += out (accum=1).
__global__ __launch_bounds__(256) void sage_wmma(
    const float* __restrict__ agg, const float* __restrict__ cnt,
    const float* __restrict__ hdst,
    const float* __restrict__ Wl, const float* __restrict__ Wr,
    const float* __restrict__ bl, float* __restrict__ hsum, int accum) {
  __shared__ unsigned As32[16 * 16];
  __shared__ unsigned Bs32[128 * 16];
  __shared__ float rinv[16];
  __shared__ float rowsq[16];
  const bf16* Asb = (const bf16*)As32;
  const bf16* Bsb = (const bf16*)Bs32;
  const int tid  = threadIdx.x;
  const int lane = tid & 31;
  const int wave = tid >> 5;
  const long long row0 = (long long)blockIdx.x * 16;
  const int am  = lane & 15;
  const int akb = (lane < 16) ? 0 : 16;
  const int bkb = (lane < 16) ? 0 : 16;
  const int col = (wave << 4) + (lane & 15);
  if (tid < 16) {
    rinv[tid]  = 1.0f / fmaxf(cnt[row0 + tid], 1.0f);
    rowsq[tid] = 0.0f;
  }
  __syncthreads();
  v8f acc = {};
#pragma unroll
  for (int pass = 0; pass < 2; ++pass) {
    const float* Asrc = pass ? hdst : agg;
    const float* Bsrc = pass ? Wr : Wl;
#pragma unroll
    for (int k0 = 0; k0 < 128; k0 += 32) {
      {
        int r = tid >> 4, kp = tid & 15;
        float2 f = *(const float2*)&Asrc[(row0 + r) * 128 + k0 + 2 * kp];
        if (!pass) { f.x *= rinv[r]; f.y *= rinv[r]; }   // mean aggregation
        As32[r * 16 + kp] = (unsigned)f2bfu(f.x) | ((unsigned)f2bfu(f.y) << 16);
      }
#pragma unroll
      for (int i = 0; i < 8; ++i) {
        int idx = tid + i * 256;
        int kh = idx >> 7, c = idx & 127;
        float f0 = Bsrc[(k0 + 2 * kh) * 128 + c];
        float f1 = Bsrc[(k0 + 2 * kh + 1) * 128 + c];
        Bs32[c * 16 + kh] = (unsigned)f2bfu(f0) | ((unsigned)f2bfu(f1) << 16);
      }
      __syncthreads();
      v16bf a, b;
#pragma unroll
      for (int e = 0; e < 16; ++e) {
        a[e] = Asb[am * 32 + akb + e];
        b[e] = Bsb[col * 32 + bkb + e];
      }
      acc = __builtin_amdgcn_wmma_f32_16x16x32_bf16(false, a, false, b,
                                                    (short)0, acc, false, false);
      __syncthreads();
    }
  }
  const int mofs = (lane < 16) ? 0 : 8;
  const float bv = bl[col];
  float vals[8];
#pragma unroll
  for (int r = 0; r < 8; ++r) {
    vals[r] = acc[r] + bv;
    atomicAdd(&rowsq[r + mofs], vals[r] * vals[r]);      // ds_add_f32
  }
  __syncthreads();
#pragma unroll
  for (int r = 0; r < 8; ++r) {
    float scale = 1.0f / fmaxf(sqrtf(rowsq[r + mofs]), 1e-12f);
    long long o = (row0 + r + mofs) * 128 + col;
    float outv = vals[r] * scale;
    hsum[o] = accum ? (hsum[o] + outv) : outv;           // HeteroConv sum
  }
}

// ---------------------------------------------------------------------------
// Wave-per-row LayerNorm + ReLU + residual: h = relu(ln(hsum)*g+b) + h.
__global__ __launch_bounds__(256) void ln_relu_res(
    const float* __restrict__ hsum, const float* __restrict__ g,
    const float* __restrict__ b, float* __restrict__ h) {
  const int lane = threadIdx.x & 31;
  const int wave = threadIdx.x >> 5;
  const long long row = (long long)blockIdx.x * 8 + wave;
  const float4 v = ((const float4*)(hsum + row * 128))[lane];
  float s = v.x + v.y + v.z + v.w;
  float q = v.x * v.x + v.y * v.y + v.z * v.z + v.w * v.w;
#pragma unroll
  for (int m = 16; m >= 1; m >>= 1) {
    s += __shfl_xor(s, m, 32);
    q += __shfl_xor(q, m, 32);
  }
  const float mean = s * (1.0f / 128.0f);
  const float var  = q * (1.0f / 128.0f) - mean * mean;
  const float rstd = rsqrtf(var + 1e-5f);
  const float4 gv = ((const float4*)g)[lane];
  const float4 bv = ((const float4*)b)[lane];
  float4 hv = ((const float4*)(h + row * 128))[lane];
  float4 o;
  o.x = fmaxf((v.x - mean) * rstd * gv.x + bv.x, 0.0f) + hv.x;
  o.y = fmaxf((v.y - mean) * rstd * gv.y + bv.y, 0.0f) + hv.y;
  o.z = fmaxf((v.z - mean) * rstd * gv.z + bv.z, 0.0f) + hv.z;
  o.w = fmaxf((v.w - mean) * rstd * gv.w + bv.w, 0.0f) + hv.w;
  ((float4*)(h + row * 128))[lane] = o;
}

// ---------------------------------------------------------------------------
extern "C" void kernel_launch(void* const* d_in, const int* in_sizes, int n_in,
                              void* d_out, int out_size, void* d_ws, size_t ws_size,
                              hipStream_t stream) {
  (void)n_in; (void)out_size; (void)ws_size;
  const int H = 128;
  const int NN[3]  = {20000, 100000, 30000};    // drug, protein, disease
  const int IND[3] = {1024, 1280, 768};

  const float* x[3] = {(const float*)d_in[0], (const float*)d_in[1], (const float*)d_in[2]};
  const int* esrc[5]; const int* edst[5]; int ecnt[5];
  for (int e = 0; e < 5; ++e) {                 // ETS order: targets, rev_targets, assoc, rev_assoc, treats
    esrc[e] = (const int*)d_in[3 + 2 * e];
    edst[e] = (const int*)d_in[4 + 2 * e];
    ecnt[e] = in_sizes[3 + 2 * e];
  }
  const int edt[5] = {1, 0, 2, 1, 2};           // dst node-type of each edge type

  // params flattened as a jax pytree (dicts in sorted-key order):
  //   layers[0..2]: conv{sorted edge keys: each Wl,Wr,bl} (15),
  //                 norm{disease b,g; drug b,g; protein b,g} (6)
  //   out:  disease W,b; drug W,b; protein W,b   (6)
  //   proj: disease, drug, protein               (3)
  const int P = 13;
  const int sortIdx[3] = {1, 2, 0};             // type t -> position among sorted {disease,drug,protein}

  // workspace carve (fp32): h[3] | hsum[3] | agg(max dst=protein) | cnt[5 edge types]
  float* ws = (float*)d_ws;
  size_t off = 0;
  float* h[3]; float* hs[3];
  for (int t = 0; t < 3; ++t) { h[t]  = ws + off; off += (size_t)NN[t] * H; }
  for (int t = 0; t < 3; ++t) { hs[t] = ws + off; off += (size_t)NN[t] * H; }
  float* agg = ws + off; off += (size_t)NN[1] * H;
  float* cntE[5];
  float* cnt0 = ws + off;
  long long cntTot = 0;
  for (int e = 0; e < 5; ++e) { cntE[e] = ws + off; off += NN[edt[e]]; cntTot += NN[edt[e]]; }

  // 0) per-edge-type degrees (layer-invariant)
  zero_f32<<<256, 256, 0, stream>>>(cnt0, cntTot);
  for (int e = 0; e < 5; ++e)
    edge_count<<<(ecnt[e] + 255) / 256, 256, 0, stream>>>(edst[e], cntE[e], ecnt[e]);

  // 1) per-type projection: h = relu(x @ Wproj)
  for (int t = 0; t < 3; ++t) {
    const float* W = (const float*)d_in[P + 69 + sortIdx[t]];
    gemm_bf16_128<<<NN[t] / 16, 256, 0, stream>>>(x[t], W, nullptr, h[t], IND[t], 1);
  }

  // conv sorted order: rev_assoc(e3,dis->prot), targets(e0,drug->prot),
  // treats(e4,drug->dis), assoc(e2,prot->dis), rev_targets(e1,prot->drug)
  struct CD { int e, st, dt; };
  const CD convs[5] = {{3, 2, 1}, {0, 0, 1}, {4, 0, 2}, {2, 1, 2}, {1, 1, 0}};

  // 2) layers
  for (int l = 0; l < 3; ++l) {
    const int base = P + l * 21;
    for (int k = 0; k < 5; ++k) {
      const CD c = convs[k];
      const int nd = NN[c.dt];
      const int E  = ecnt[c.e];
      const int accum = !(k == 0 || k == 2 || k == 4);  // first conv per dst stores
      zero_f32<<<2048, 256, 0, stream>>>(agg, (long long)nd * H);
      sage_scatter<<<(E * 32 + 255) / 256, 256, 0, stream>>>(
          h[c.st], esrc[c.e], edst[c.e], agg, E);
      sage_wmma<<<nd / 16, 256, 0, stream>>>(
          agg, cntE[c.e], h[c.dt],
          (const float*)d_in[base + k * 3 + 0],   // Wl
          (const float*)d_in[base + k * 3 + 1],   // Wr
          (const float*)d_in[base + k * 3 + 2],   // bl
          hs[c.dt], accum);
    }
    const int nbase = base + 15;
    for (int t = 0; t < 3; ++t) {
      const float* bb = (const float*)d_in[nbase + sortIdx[t] * 2 + 0];
      const float* gg = (const float*)d_in[nbase + sortIdx[t] * 2 + 1];
      ln_relu_res<<<NN[t] / 8, 256, 0, stream>>>(hs[t], gg, bb, h[t]);
    }
  }

  // 3) per-type output heads into d_out (drug | protein | disease)
  float* out = (float*)d_out;
  const size_t oofs[3] = {0, (size_t)NN[0] * H, (size_t)(NN[0] + NN[1]) * H};
  for (int t = 0; t < 3; ++t) {
    const float* W  = (const float*)d_in[P + 63 + sortIdx[t] * 2 + 0];
    const float* bb = (const float*)d_in[P + 63 + sortIdx[t] * 2 + 1];
    gemm_bf16_128<<<NN[t] / 16, 256, 0, stream>>>(h[t], W, bb, out + oofs[t], H, 0);
  }
}
